// Network_9251359556128
// MI455X (gfx1250) — matrix operand order repaired and checked by reading, not compiled, not run on hardware
//
#include <hip/hip_runtime.h>
#include <hip/hip_bf16.h>
#include <math.h>

typedef __bf16 bf16;
typedef bf16 v16bf __attribute__((ext_vector_type(16)));
typedef float v8f  __attribute__((ext_vector_type(8)));

union Frag16 { v16bf v; bf16 e[16]; unsigned u[8]; };

__device__ __forceinline__ float mishf(float x) {
  float sp = (x > 20.0f) ? x : log1pf(__expf(x));
  return x * tanhf(sp);
}

// ---------------- weight packing: (Cout,Cin,KH,KW) f32 -> (Cout,Kpad) bf16 ----
__global__ void k_pack_w(const float* __restrict__ w, bf16* __restrict__ dst,
                         int Cout, int Ktot, int Kpad) {
  int i = blockIdx.x * blockDim.x + threadIdx.x;
  if (i >= Cout * Kpad) return;
  int co = i / Kpad, k = i - co * Kpad;
  float v = (k < Ktot) ? w[(long)co * Ktot + k] : 0.0f;
  dst[i] = (bf16)v;
}

__global__ void k_zero(float* __restrict__ p, int n) {
  int i = blockIdx.x * blockDim.x + threadIdx.x;
  if (i < n) p[i] = 0.0f;
}

// ---------------- implicit-GEMM conv via v_wmma_f32_16x16x32_bf16 ------------
// One wave computes a (16*NCO Cout) x (32 pixel) f32 macro-tile: per K-chunk
// the two B (activation) fragments are built once and reused by NCO weight
// fragments -> 2*NCO WMMAs per chunk, 4x less im2col gather per matrix op.
// postop: 0 = +bias, 1 = +bias then mish, 2 = +bias then accumulate into out.
template <int KH, int KW, int NCO>
__global__ __launch_bounds__(256) void k_conv_wmma(
    const float* __restrict__ in, const bf16* __restrict__ Wp,
    const float* __restrict__ bias, float* __restrict__ out,
    int B, int Cin, int H, int W,
    int Cout, int CoutTot, int co_off,
    int pad, int Ktot, int Kpad, int postop) {
  const unsigned lane = __builtin_amdgcn_mbcnt_lo(~0u, 0u); // wave32 lane id
  const int wave = threadIdx.x >> 5;
  const int tilesX = W >> 5, nGrp = Cout / (16 * NCO);
  long tile = (long)blockIdx.x * (blockDim.x >> 5) + wave;  // wave-uniform
  long total = (long)B * H * tilesX * nGrp;
  if (tile >= total) return;                                // whole wave exits

  int grp = (int)(tile % nGrp); long t = tile / nGrp;
  int tx = (int)(t % tilesX); t /= tilesX;
  int y  = (int)(t % H);      int nb = (int)(t / H);
  const int x0 = tx << 5, co_base = grp * 16 * NCO;

  const int m  = lane & 15;           // A row / B column / pixel-in-tile
  const int hi = (lane >> 4) & 1;
  const int aOff = hi ? 16 : 0;       // A: lanes 16-31 hold K+16
  const int bOff = hi ? 16 : 0;       // B: lanes 16-31 hold K+16
  constexpr int KHW = KH * KW;        // compile-time: div-by-const is cheap
  const long inNB = (long)nb * Cin * H * W;

  v8f acc[NCO][2];
  #pragma unroll
  for (int j = 0; j < NCO; ++j)
    #pragma unroll
    for (int i = 0; i < 8; ++i) { acc[j][0][i] = 0.0f; acc[j][1][i] = 0.0f; }

  const unsigned* wrow = (const unsigned*)(Wp + (long)(co_base + m) * Kpad);
  const long wstride = (long)8 * Kpad;             // 16 rows, in dwords

  auto kstep = [&](int kb, bool tail) {
    // B fragments (two 16-pixel columns), f32 -> bf16, implicit im2col
    Frag16 B0, B1;
    #pragma unroll
    for (int h = 0; h < 16; ++h) {
      int k = kb + bOff + h;
      float v0 = 0.0f, v1 = 0.0f;
      if (!tail || k < Ktot) {
        int ci = k / KHW, rr = k - ci * KHW;
        int ky = rr / KW, kx = rr - ky * KW;
        int iy = y + ky - pad;
        int ix0 = x0 + m + kx - pad;
        int ix1 = ix0 + 16;
        const float* row = in + inNB + ((long)ci * H + iy) * W;
        bool okY = (unsigned)iy < (unsigned)H;
        if (okY && (unsigned)ix0 < (unsigned)W) v0 = row[ix0];
        if (okY && (unsigned)ix1 < (unsigned)W) v1 = row[ix1];
      }
      B0.e[h] = (bf16)v0;
      B1.e[h] = (bf16)v1;
    }
    __builtin_prefetch(wrow + ((kb + 32) >> 1), 0, 1); // next weight chunk
    // NCO weight fragments reuse B0/B1
    #pragma unroll
    for (int j = 0; j < NCO; ++j) {
      Frag16 A;
      const unsigned* wj = wrow + j * wstride;
      #pragma unroll
      for (int v = 0; v < 8; ++v) {
        int h = 2 * v;
        int kl = h + aOff;
        A.u[v] = wj[(kb + kl) >> 1];
      }
      acc[j][0] = __builtin_amdgcn_wmma_f32_16x16x32_bf16(
          false, A.v, false, B0.v, (short)0, acc[j][0], false, false);
      acc[j][1] = __builtin_amdgcn_wmma_f32_16x16x32_bf16(
          false, A.v, false, B1.v, (short)0, acc[j][1], false, false);
    }
  };

  int kb = 0;
  for (; kb + 32 <= Ktot; kb += 32) kstep(kb, false); // check-free main loop
  for (; kb < Kpad; kb += 32) kstep(kb, true);        // padded tail chunk

  const int xo0 = x0 + m, xo1 = xo0 + 16;
  #pragma unroll
  for (int j = 0; j < NCO; ++j) {
    #pragma unroll
    for (int r = 0; r < 8; ++r) {
      int co = co_base + j * 16 + r + (hi ? 8 : 0); // C/D layout: M=r / M=8+r
      float bv = bias[co];
      float v0 = acc[j][0][r] + bv, v1 = acc[j][1][r] + bv;
      long row = (((long)nb * CoutTot + (co_off + co)) * H + y) * W;
      long oi0 = row + xo0, oi1 = row + xo1;
      if (postop == 1) { v0 = mishf(v0); v1 = mishf(v1); }
      else if (postop == 2) { v0 += out[oi0]; v1 += out[oi1]; }
      out[oi0] = v0;
      out[oi1] = v1;
    }
  }
}

// ---------------- per-channel batch statistics (sum, sumsq) ------------------
__global__ __launch_bounds__(256) void k_bn_stats(
    const float* __restrict__ x, float* __restrict__ st,
    int B, int C, long HW) {
  int c = blockIdx.x;
  long per = (long)B * HW;
  float s = 0.0f, s2 = 0.0f;
  for (long e = (long)blockIdx.y * blockDim.x + threadIdx.x; e < per;
       e += (long)gridDim.y * blockDim.x) {
    long nb = e / HW, p = e - nb * HW;
    float v = x[(nb * C + c) * HW + p];
    s += v; s2 += v * v;
  }
  __shared__ float sh[256], sh2[256];
  sh[threadIdx.x] = s; sh2[threadIdx.x] = s2;
  __syncthreads();
  for (int o = 128; o > 0; o >>= 1) {
    if ((int)threadIdx.x < o) {
      sh[threadIdx.x] += sh[threadIdx.x + o];
      sh2[threadIdx.x] += sh2[threadIdx.x + o];
    }
    __syncthreads();
  }
  if (threadIdx.x == 0) {
    atomicAdd(&st[c], sh[0]);
    atomicAdd(&st[C + c], sh2[0]);
  }
}

// ---------------- BN normalize + mish (optionally teed to second dst) --------
__global__ void k_bn_mish(const float* __restrict__ x, float* __restrict__ y,
                          float* __restrict__ y2, const float* __restrict__ st,
                          const float* __restrict__ g, const float* __restrict__ b,
                          int C, long HW, long total, float invN) {
  long i = (long)blockIdx.x * blockDim.x + threadIdx.x;
  if (i >= total) return;
  int c = (int)((i / HW) % C);
  float mean = st[c] * invN;
  float var  = st[C + c] * invN - mean * mean;
  float v = (x[i] - mean) * rsqrtf(var + 1e-5f) * g[c] + b[c];
  v = mishf(v);
  y[i] = v;
  if (y2) y2[i] = v;
}

// ---------------- per-(n,c) spatial sum for SE attention ---------------------
__global__ __launch_bounds__(256) void k_chan_sum(
    const float* __restrict__ x, float* __restrict__ dst, long HW) {
  long bc = blockIdx.x;
  float s = 0.0f;
  for (long p = (long)blockIdx.y * blockDim.x + threadIdx.x; p < HW;
       p += (long)gridDim.y * blockDim.x)
    s += x[bc * HW + p];
  __shared__ float sh[256];
  sh[threadIdx.x] = s;
  __syncthreads();
  for (int o = 128; o > 0; o >>= 1) {
    if ((int)threadIdx.x < o) sh[threadIdx.x] += sh[threadIdx.x + o];
    __syncthreads();
  }
  if (threadIdx.x == 0) atomicAdd(&dst[bc], sh[0]);
}

// ---------------- FC + sigmoid -> channel scale ------------------------------
__global__ void k_att_fc(const float* __restrict__ asum, const float* __restrict__ fw,
                         const float* __restrict__ fb, float* __restrict__ scale,
                         int B, int C, float invHW) {
  int i = blockIdx.x * blockDim.x + threadIdx.x;
  if (i >= B * C) return;
  int nb = i / C, co = i - nb * C;
  float s = fb[co];
  for (int c = 0; c < C; ++c)
    s += (asum[nb * C + c] * invHW) * fw[co * C + c];
  scale[i] = 1.0f / (1.0f + __expf(-s));
}

// ---------------- out = sigmoid(s) * h + residual ----------------------------
__global__ void k_scale_res(const float* __restrict__ h, const float* __restrict__ res,
                            const float* __restrict__ scale, float* __restrict__ o,
                            int C, long HW, long total) {
  long i = (long)blockIdx.x * blockDim.x + threadIdx.x;
  if (i >= total) return;
  long nc = i / HW;
  int c = (int)(nc % C), nb = (int)(nc / C);
  o[i] = scale[nb * C + c] * h[i] + res[i];
}

// ---------------- Haar DWT: cA -> out, (cH,cV,cD) -> 3C-channel tensor -------
__global__ void k_haar(const float* __restrict__ src, float* __restrict__ cA,
                       float* __restrict__ D, int B, int C, int H, int W) {
  int Ho = H >> 1, Wo = W >> 1;
  long total = (long)B * C * Ho * Wo;
  long i = (long)blockIdx.x * blockDim.x + threadIdx.x;
  if (i >= total) return;
  int xo = (int)(i % Wo); long t = i / Wo;
  int yo = (int)(t % Ho); t /= Ho;
  int c = (int)(t % C);   int nb = (int)(t / C);
  const float* p = src + (((long)nb * C + c) * H + 2 * yo) * W + 2 * xo;
  float a = p[0], b = p[1], cc = p[W], d = p[W + 1];
  cA[i] = (a + b + cc + d) * 0.5f;
  long hwo = (long)Ho * Wo, pix = (long)yo * Wo + xo;
  long base = (long)nb * 3 * C * hwo;
  D[base + (long)c * hwo + pix]           = (a + b - cc - d) * 0.5f;
  D[base + (long)(C + c) * hwo + pix]     = (a - b + cc - d) * 0.5f;
  D[base + (long)(2 * C + c) * hwo + pix] = (a - b - cc + d) * 0.5f;
}

// =============================================================================
extern "C" void kernel_launch(void* const* d_in, const int* in_sizes, int n_in,
                              void* d_out, int out_size, void* d_ws, size_t ws_size,
                              hipStream_t stream) {
  (void)in_sizes; (void)n_in; (void)out_size; (void)ws_size;
  const float* x        = (const float*)d_in[0];
  const float* w_res    = (const float*)d_in[1];
  const float* b_res    = (const float*)d_in[2];
  const float* w_c1     = (const float*)d_in[3];
  const float* b_c1     = (const float*)d_in[4];
  const float* bn1_g    = (const float*)d_in[5];
  const float* bn1_b    = (const float*)d_in[6];
  const float* w_c2     = (const float*)d_in[7];
  const float* b_c2     = (const float*)d_in[8];
  const float* bn2_g    = (const float*)d_in[9];
  const float* bn2_b    = (const float*)d_in[10];
  const float* w_att    = (const float*)d_in[11];
  const float* b_att    = (const float*)d_in[12];
  const float* bnatt_g  = (const float*)d_in[13];
  const float* bnatt_b  = (const float*)d_in[14];
  const float* fc_w     = (const float*)d_in[15];
  const float* fc_b     = (const float*)d_in[16];
  const float* w_p3     = (const float*)d_in[17];
  const float* b_p3     = (const float*)d_in[18];
  const float* w_p5     = (const float*)d_in[19];
  const float* b_p5     = (const float*)d_in[20];
  const float* w_p7     = (const float*)d_in[21];
  const float* b_p7     = (const float*)d_in[22];
  const float* w_skip   = (const float*)d_in[23];
  const float* b_skip   = (const float*)d_in[24];
  const float* w_split  = (const float*)d_in[25];
  const float* b_split  = (const float*)d_in[26];
  const float* w_high   = (const float*)d_in[27];
  const float* b_high   = (const float*)d_in[28];
  const float* bnh_g    = (const float*)d_in[29];
  const float* bnh_b    = (const float*)d_in[30];
  float* out = (float*)d_out;

  const int B = 8, CIN = 3, C = 64, H = 256, W = 256;
  const long HW = (long)H * W;
  const long N64 = (long)B * C * HW;        // 33,554,432
  const long N16 = (long)B * 16 * HW;       //  8,388,608
  const long NP  = (long)B * 3 * C * HW;    // 100,663,296
  const int Ho = H / 2, Wo = W / 2;
  const long NLOW = (long)B * C * Ho * Wo;  //  8,388,608

  // ---- workspace layout (floats), with lifetime-based aliasing ----
  float* ws = (float*)d_ws;
  size_t off = 0;
  float* stats = ws + off; off += 2048;       // bn1@0 bn2@32 bnatt@160 bnh@288 asum@416 scale@928
  bf16*  wpBase = (bf16*)(ws + off); off += 196608;  // 393,216 bf16 slots
  float* bufA = ws + off; off += N64;  // residual; later (cH,cV,cD) (B,192,128,128)
  float* bufB = ws + off; off += N64;  // h;        later split-conv output
  float* bufC = ws + off; off += N64;  // a;        later SE-scaled out
  float* bufD = ws + off; off += N16;  // 16-ch t1; later high-conv output
  float* bufP = ws + off; off += NP;   // pyramid concat + skip

  // zero the atomic accumulators
  k_zero<<<(2048 + 255) / 256, 256, 0, stream>>>(stats, 2048);

  // ---- pack all weights to bf16 GEMM layout ----
  struct WP { const bf16* p; int Ktot, Kpad; };
  size_t wo = 0;
  auto pack = [&](const float* w, int Cout, int Cin, int KH, int KW) -> WP {
    int Ktot = Cin * KH * KW, Kp = (Ktot + 31) & ~31;
    bf16* dst = wpBase + wo; wo += (size_t)Cout * Kp;
    int n = Cout * Kp;
    k_pack_w<<<(n + 255) / 256, 256, 0, stream>>>(w, dst, Cout, Ktot, Kp);
    return {dst, Ktot, Kp};
  };
  WP wpRes   = pack(w_res,   64,  CIN, 1, 1);
  WP wpC1    = pack(w_c1,    16,  CIN, 3, 3);
  WP wpC2    = pack(w_c2,    64,  16,  3, 3);
  WP wpAtt   = pack(w_att,   64,  64,  1, 1);
  WP wpP3    = pack(w_p3,    64,  64,  3, 3);
  WP wpP5    = pack(w_p5,    64,  64,  5, 5);
  WP wpP7    = pack(w_p7,    64,  64,  7, 7);
  WP wpSkip  = pack(w_skip,  192, CIN, 1, 1);
  WP wpSplit = pack(w_split, 64,  192, 1, 1);
  WP wpHigh  = pack(w_high,  64,  192, 1, 1);

  // NCO=4 (64 Cout rows per wave) when Cout % 64 == 0, else NCO=1 (16 rows).
  auto conv = [&](const float* in, WP wp, const float* bias, float* o,
                  int Cin, int Cout, int CoutTot, int co_off,
                  int KH, int pad, int Hh, int Ww, int postop) {
    bool big = (Cout % 64) == 0;
    int rowsPerWave = big ? 64 : 16;
    long total = (long)B * Hh * (Ww >> 5) * (Cout / rowsPerWave);
    int blocks = (int)((total + 7) / 8);     // 8 waves / 256-thread block
#define LAUNCH(KH_, KW_, NCO_)                                                \
    k_conv_wmma<KH_, KW_, NCO_><<<blocks, 256, 0, stream>>>(                  \
        in, wp.p, bias, o, B, Cin, Hh, Ww, Cout, CoutTot, co_off,             \
        pad, wp.Ktot, wp.Kpad, postop)
    if (KH == 1)      LAUNCH(1, 1, 4);
    else if (KH == 3) { if (big) LAUNCH(3, 3, 4); else LAUNCH(3, 3, 1); }
    else if (KH == 5) LAUNCH(5, 5, 4);
    else              LAUNCH(7, 7, 4);
#undef LAUNCH
  };
  auto bn = [&](const float* buf, float* st, const float* g, const float* bb,
                int Cc, int Hh, int Ww, float* y, float* y2) {
    long hw = (long)Hh * Ww;
    dim3 grid(Cc, 64);
    k_bn_stats<<<grid, 256, 0, stream>>>(buf, st, B, Cc, hw);
    long total = (long)B * Cc * hw;
    k_bn_mish<<<(int)((total + 255) / 256), 256, 0, stream>>>(
        buf, y, y2, st, g, bb, Cc, hw, total, 1.0f / (float)(B * hw));
  };

  // residual = conv1x1(x)                                  -> bufA
  conv(x, wpRes, b_res, bufA, CIN, 64, 64, 0, 1, 0, H, W, 0);
  // t1 = conv3x3(x) ; bn+mish                              -> bufD
  conv(x, wpC1, b_c1, bufD, CIN, 16, 16, 0, 3, 1, H, W, 0);
  bn(bufD, stats + 0, bn1_g, bn1_b, 16, H, W, bufD, nullptr);
  // h = conv3x3(t1) ; bn+mish (tee conn_x into d_out)      -> bufB
  conv(bufD, wpC2, b_c2, bufB, 16, 64, 64, 0, 3, 1, H, W, 0);
  bn(bufB, stats + 32, bn2_g, bn2_b, 64, H, W, bufB, out + 2 * NLOW);
  // a = conv1x1(h) ; bn+mish                               -> bufC
  conv(bufB, wpAtt, b_att, bufC, 64, 64, 64, 0, 1, 0, H, W, 0);
  bn(bufC, stats + 160, bnatt_g, bnatt_b, 64, H, W, bufC, nullptr);
  // SE: spatial mean -> fc -> sigmoid -> scale*h + residual -> bufC
  { dim3 grid(B * C, 16);
    k_chan_sum<<<grid, 256, 0, stream>>>(bufC, stats + 416, HW); }
  k_att_fc<<<2, 256, 0, stream>>>(stats + 416, fc_w, fc_b, stats + 928,
                                  B, C, 1.0f / (float)HW);
  k_scale_res<<<(int)((N64 + 255) / 256), 256, 0, stream>>>(
      bufB, bufA, stats + 928, bufC, C, HW, N64);
  // pyramid (mish fused) into concat channels of bufP, then skip accumulates
  conv(bufC, wpP3, b_p3, bufP, 64, 64, 192, 0,   3, 1, H, W, 1);
  conv(bufC, wpP5, b_p5, bufP, 64, 64, 192, 64,  5, 2, H, W, 1);
  conv(bufC, wpP7, b_p7, bufP, 64, 64, 192, 128, 7, 3, H, W, 1);
  conv(x, wpSkip, b_skip, bufP, CIN, 192, 192, 0, 1, 0, H, W, 2);
  // split: conv1x1 192->64                                 -> bufB
  conv(bufP, wpSplit, b_split, bufB, 192, 64, 64, 0, 1, 0, H, W, 0);
  // Haar DWT: cA -> d_out[0:], (cH,cV,cD) as 192ch         -> bufA
  k_haar<<<(int)((NLOW + 255) / 256), 256, 0, stream>>>(bufB, out, bufA, B, C, H, W);
  // x_high = mish(bn(conv1x1(highbands)))                  -> d_out[NLOW:]
  conv(bufA, wpHigh, b_high, bufD, 192, 64, 64, 0, 1, 0, Ho, Wo, 0);
  bn(bufD, stats + 288, bnh_g, bnh_b, 64, Ho, Wo, out + NLOW, nullptr);
}
